// MaeContrastiveLoss_42786464202873
// MI455X (gfx1250) — matrix operand order repaired and checked by reading, not compiled, not run on hardware
//
#include <hip/hip_runtime.h>
#include <hip/hip_fp16.h>

// MI455X / gfx1250, wave32, WMMA f16 -> f32 accumulate, LDS-staged GEMM tiles.

typedef __attribute__((ext_vector_type(16))) _Float16 v16h;
typedef __attribute__((ext_vector_type(8)))  _Float16 v8h;
typedef __attribute__((ext_vector_type(8)))  float    v8f;
typedef __attribute__((ext_vector_type(4)))  float    v4f;

#define NB   512      // batch
#define NL   128      // seq length (1 anchor + 127 rest)
#define ND   256      // feature dim
#define NLP  128      // rest rows per sample after zero-padding (127 real + 1 zero)
#define MT   128      // block tile in M (anchors per block)
#define LSTR 264      // padded LDS row stride in halves (528B = 132 dw = 4 mod 64 banks)

// ---------------------------------------------------------------------------
// Pass 1: L2-normalize each (b,l) row, convert to f16.
//   row l==0  -> anchors A[b][:], also zero-pad Rp[b][127][:]
//   row l>=1  -> Rp[b][l-1][:]
// One wave per row: 32 lanes x 8 floats = 256.
// ---------------------------------------------------------------------------
__global__ __launch_bounds__(256)
void k_normalize(const float* __restrict__ in,
                 _Float16* __restrict__ Aa,
                 _Float16* __restrict__ Rp)
{
    const int lane = threadIdx.x & 31;
    const int row  = blockIdx.x * 8 + (threadIdx.x >> 5);   // 0 .. NB*NL-1
    const int b = row >> 7;
    const int l = row & 127;

    const float* p = in + (size_t)row * ND + lane * 8;
    v4f x0 = *(const v4f*)(p);
    v4f x1 = *(const v4f*)(p + 4);

    float xv[8];
    #pragma unroll
    for (int i = 0; i < 4; ++i) { xv[i] = x0[i]; xv[4 + i] = x1[i]; }

    float ss = 0.f;
    #pragma unroll
    for (int i = 0; i < 8; ++i) ss += xv[i] * xv[i];

    #pragma unroll
    for (int m = 16; m >= 1; m >>= 1) ss += __shfl_xor(ss, m, 32);

    const float scale = 1.0f / fmaxf(sqrtf(ss), 1e-12f);

    v8h h;
    #pragma unroll
    for (int i = 0; i < 8; ++i) h[i] = (_Float16)(xv[i] * scale);

    if (l == 0) {
        *(v8h*)(Aa + (size_t)b * ND + lane * 8) = h;
        v8h z;
        #pragma unroll
        for (int i = 0; i < 8; ++i) z[i] = (_Float16)0.0f;
        *(v8h*)(Rp + ((size_t)b * NLP + (NLP - 1)) * ND + lane * 8) = z;
    } else {
        *(v8h*)(Rp + ((size_t)b * NLP + (l - 1)) * ND + lane * 8) = h;
    }
}

// ---------------------------------------------------------------------------
// Pass 2: fused GEMM + exp + K-group reduction, LDS-staged.
// Grid: (512/128, 512). Block tile: 128 anchors x 128 rest rows (= one j).
// Stage A(128x256 f16) + B(128x256 f16) to LDS once (132KB), then 8 waves:
// wave w -> m-sub (w&3)*32 (2 WMMA rows), n-half (w>>2)*64 (4 WMMA cols).
// K = 256 in 8 steps of 32 -> 64 v_wmma per wave, fragments double-buffered.
// ---------------------------------------------------------------------------
__global__ __launch_bounds__(256)
void k_gemm_exp(const _Float16* __restrict__ Aa,
                const _Float16* __restrict__ Rp,
                float* __restrict__ E)
{
    extern __shared__ _Float16 lds[];
    _Float16* Als = lds;                        // [MT][LSTR]
    _Float16* Bls = lds + (size_t)MT * LSTR;    // [NLP][LSTR]
    float* smPartial = (float*)(lds + 2 * (size_t)MT * LSTR);  // [2][MT]

    const int tid    = threadIdx.x;
    const int lane   = tid & 31;
    const int wid    = tid >> 5;     // 0..7
    const int mSub   = wid & 3;      // 0..3 -> 32-row chunk
    const int nHalf  = wid >> 2;     // 0..1 -> 64-col chunk
    const int j      = blockIdx.y;
    const int laneLo = lane & 15;
    const int laneHi = lane >> 4;

    // ---- stage A and B tiles into LDS (2 threads per row, 256B chunks) ----
    {
        const int r = tid >> 1;      // 0..127
        const int h = tid & 1;       // half of the 256-element row
        const _Float16* gA = Aa + ((size_t)(blockIdx.x * MT + r)) * ND + h * 128;
        const _Float16* gB = Rp + ((size_t)(j * NLP + r)) * ND + h * 128;
        _Float16* lA = Als + (size_t)r * LSTR + h * 128;
        _Float16* lB = Bls + (size_t)r * LSTR + h * 128;
        #pragma unroll
        for (int i = 0; i < 16; ++i) {
            *(v8h*)(lA + i * 8) = *(const v8h*)(gA + i * 8);
            *(v8h*)(lB + i * 8) = *(const v8h*)(gB + i * 8);
        }
    }
    __syncthreads();

    // Fragment bases in LDS.
    // A fragment (16x32): lane group g holds K in {kb..kb+7, kb+16..kb+23}, kb=g*8.
    // B fragment (32x16): lane group g holds K in {kb..kb+7, kb+16..kb+23}, kb=g*8.
    const _Float16* aBase[2];
    #pragma unroll
    for (int a = 0; a < 2; ++a)
        aBase[a] = Als + (size_t)(mSub * 32 + a * 16 + laneLo) * LSTR;
    const _Float16* bBase[4];
    #pragma unroll
    for (int t = 0; t < 4; ++t)
        bBase[t] = Bls + (size_t)(nHalf * 64 + t * 16 + laneLo) * LSTR;
    const int aK0 = laneHi * 8;
    const int bK0 = laneHi * 8;

    auto loadA = [&](const _Float16* base, int k0) -> v16h {
        v8h lo = *(const v8h*)(base + k0 + aK0);
        v8h hi = *(const v8h*)(base + k0 + aK0 + 16);
        v16h f;
        #pragma unroll
        for (int i = 0; i < 8; ++i) { f[i] = lo[i]; f[8 + i] = hi[i]; }
        return f;
    };
    auto loadB = [&](const _Float16* base, int k0) -> v16h {
        v8h lo = *(const v8h*)(base + k0 + bK0);
        v8h hi = *(const v8h*)(base + k0 + bK0 + 16);
        v16h f;
        #pragma unroll
        for (int i = 0; i < 8; ++i) { f[i] = lo[i]; f[8 + i] = hi[i]; }
        return f;
    };

    v8f acc[2][4];
    #pragma unroll
    for (int a = 0; a < 2; ++a)
        #pragma unroll
        for (int t = 0; t < 4; ++t)
            #pragma unroll
            for (int r = 0; r < 8; ++r) acc[a][t][r] = 0.0f;

    // Double-buffered K loop: ds_loads for step kk+1 issue before WMMAs of kk.
    v16h aCur[2], bCur[4];
    #pragma unroll
    for (int a = 0; a < 2; ++a) aCur[a] = loadA(aBase[a], 0);
    #pragma unroll
    for (int t = 0; t < 4; ++t) bCur[t] = loadB(bBase[t], 0);

    #pragma unroll
    for (int kk = 0; kk < 8; ++kk) {
        v16h aNxt[2], bNxt[4];
        if (kk < 7) {
            const int k1 = (kk + 1) * 32;
            #pragma unroll
            for (int a = 0; a < 2; ++a) aNxt[a] = loadA(aBase[a], k1);
            #pragma unroll
            for (int t = 0; t < 4; ++t) bNxt[t] = loadB(bBase[t], k1);
        }
        #pragma unroll
        for (int a = 0; a < 2; ++a)
            #pragma unroll
            for (int t = 0; t < 4; ++t)
                acc[a][t] = __builtin_amdgcn_wmma_f32_16x16x32_f16(
                    false, aCur[a], false, bCur[t], (short)0, acc[a][t],
                    false, false);
        if (kk < 7) {
            #pragma unroll
            for (int a = 0; a < 2; ++a) aCur[a] = aNxt[a];
            #pragma unroll
            for (int t = 0; t < 4; ++t) bCur[t] = bNxt[t];
        }
    }

    // C/D layout: lane l, VGPR r -> M = r + 8*(l>>4), N = (l&15).
    // Fused exp + row sum over the wave's 64 columns.
    float rowSum[2][8];
    #pragma unroll
    for (int a = 0; a < 2; ++a) {
        #pragma unroll
        for (int r = 0; r < 8; ++r) {
            float s = __expf(acc[a][0][r]) + __expf(acc[a][1][r]) +
                      __expf(acc[a][2][r]) + __expf(acc[a][3][r]);
            s += __shfl_xor(s, 1, 32);
            s += __shfl_xor(s, 2, 32);
            s += __shfl_xor(s, 4, 32);
            s += __shfl_xor(s, 8, 32);
            rowSum[a][r] = s;
        }
    }

    if (laneLo == 0) {
        #pragma unroll
        for (int a = 0; a < 2; ++a) {
            const int mLocal = mSub * 32 + a * 16 + laneHi * 8;
            #pragma unroll
            for (int r = 0; r < 8; ++r)
                smPartial[nHalf * MT + mLocal + r] = rowSum[a][r];
        }
    }
    __syncthreads();

    if (tid < MT) {
        const int i = blockIdx.x * MT + tid;
        E[(size_t)i * NB + j] = smPartial[tid] + smPartial[MT + tid];
    }
}

// ---------------------------------------------------------------------------
// Pass 3: per-anchor reduction over j with label mask + exact pad correction.
// Padding added exp(0)=1 once per (i,j): tot -= 512, pos -= count_same(i).
// ---------------------------------------------------------------------------
__global__ __launch_bounds__(256)
void k_row_loss(const float* __restrict__ E,
                const long long* __restrict__ label,
                float* __restrict__ lossbuf)
{
    __shared__ float sTot[256], sPos[256], sCnt[256];
    const int i = blockIdx.x;
    const long long li = label[i];

    float tot = 0.f, pos = 0.f, cnt = 0.f;
    for (int j = threadIdx.x; j < NB; j += 256) {
        const float e = E[(size_t)i * NB + j];
        tot += e;
        if (label[j] == li) { pos += e; cnt += 1.f; }
    }
    sTot[threadIdx.x] = tot;
    sPos[threadIdx.x] = pos;
    sCnt[threadIdx.x] = cnt;
    __syncthreads();

    for (int s = 128; s > 0; s >>= 1) {
        if (threadIdx.x < s) {
            sTot[threadIdx.x] += sTot[threadIdx.x + s];
            sPos[threadIdx.x] += sPos[threadIdx.x + s];
            sCnt[threadIdx.x] += sCnt[threadIdx.x + s];
        }
        __syncthreads();
    }

    if (threadIdx.x == 0) {
        const float totT = sTot[0] - (float)NB;   // remove 512 pad contributions
        const float posT = sPos[0] - sCnt[0];     // remove count_same pad contributions
        lossbuf[i] = __logf(totT) - __logf(posT);
    }
}

// ---------------------------------------------------------------------------
// Pass 4: mean of 512 per-row losses (deterministic tree reduction).
// ---------------------------------------------------------------------------
__global__ __launch_bounds__(256)
void k_final(const float* __restrict__ lossbuf, float* __restrict__ out)
{
    __shared__ float s[256];
    s[threadIdx.x] = lossbuf[threadIdx.x] + lossbuf[threadIdx.x + 256];
    __syncthreads();
    for (int st = 128; st > 0; st >>= 1) {
        if (threadIdx.x < st) s[threadIdx.x] += s[threadIdx.x + st];
        __syncthreads();
    }
    if (threadIdx.x == 0) out[0] = s[0] * (1.0f / (float)NB);
}

// ---------------------------------------------------------------------------
extern "C" void kernel_launch(void* const* d_in, const int* in_sizes, int n_in,
                              void* d_out, int out_size, void* d_ws, size_t ws_size,
                              hipStream_t stream)
{
    const float*     in    = (const float*)d_in[0];      // (512,128,256) fp32
    const long long* label = (const long long*)d_in[1];  // (512,) int64
    float*           out   = (float*)d_out;              // scalar fp32

    char* ws = (char*)d_ws;
    const size_t offA  = 0;                                   // 512*256*2     = 256 KB
    const size_t offR  = offA + (size_t)NB * ND * 2;          // 512*128*256*2 = 32 MB
    const size_t offE  = offR + (size_t)NB * NLP * ND * 2;    // 512*512*4     = 1 MB
    const size_t offLb = offE + (size_t)NB * NB * 4;          // 512*4         = 2 KB

    _Float16* Aa      = (_Float16*)(ws + offA);
    _Float16* Rp      = (_Float16*)(ws + offR);
    float*    E       = (float*)(ws + offE);
    float*    lossbuf = (float*)(ws + offLb);

    // Pass 1: normalize + f16 convert + zero-pad
    k_normalize<<<(NB * NL) / 8, 256, 0, stream>>>(in, Aa, Rp);

    // Pass 2: LDS-staged WMMA GEMM + exp + per-j reduction
    dim3 g2(NB / MT, NB);
    const size_t shm = 2 * (size_t)MT * LSTR * sizeof(_Float16) + 2 * MT * sizeof(float);
    k_gemm_exp<<<g2, 256, shm, stream>>>(Aa, Rp, E);

    // Pass 3: masked row reductions
    k_row_loss<<<NB, 256, 0, stream>>>(E, label, lossbuf);

    // Pass 4: mean
    k_final<<<1, 256, 0, stream>>>(lossbuf, out);
}
